// ScalableCritic_16140487098427
// MI455X (gfx1250) — matrix operand order repaired and checked by reading, not compiled, not run on hardware
//
#include <hip/hip_runtime.h>
#include <cmath>

typedef _Float16 v8h  __attribute__((ext_vector_type(8)));
typedef _Float16 v16h __attribute__((ext_vector_type(16)));
typedef float    v8f  __attribute__((ext_vector_type(8)));

#define NEGV (-1.0e9f)

__device__ __forceinline__ float leaky02(float x) { return x >= 0.0f ? x : 0.2f * x; }

__device__ __forceinline__ void atomic_fmax(float* addr, float v) {
    __hip_atomic_fetch_max(addr, v, __ATOMIC_RELAXED, __HIP_MEMORY_SCOPE_AGENT);
}

// ---------------------------------------------------------------- K0: init
__global__ __launch_bounds__(256) void seg_init_kernel(float* __restrict__ seg, int n) {
    int i = blockIdx.x * 256 + threadIdx.x;
    if (i < n) seg[i] = NEGV;
}

// ---------------------------------------------------------------- K1: point net + segment max
// block = 256 threads = 8 waves; each wave processes 16-muon tiles.
// B matrix (w2) is swizzled to WMMA fragment layout in LDS ONCE, then held in
// 16 v16h registers for the whole kernel -> inner loop is pure WMMA.
__global__ __launch_bounds__(256) void point_kernel(
    const float* __restrict__ muons,   // [M,3]
    const int*   __restrict__ bidx,    // [M] sorted
    const float* __restrict__ cond,    // [B,4]
    const float* __restrict__ w1,      // [7,64]
    const float* __restrict__ b1,      // [64]
    const float* __restrict__ w2,      // [64,128]
    const float* __restrict__ b2,      // [128]
    float*       __restrict__ seg,     // [B,128] pre-init to -1e9
    int M, int nTiles, int rounds)
{
    // w2 in f16, pre-swizzled into WMMA B-fragment layout, contiguous k order (k = 16h + s), matching A:
    // frag = (k>>5)*8 + (c>>4); lane = (c&15) | (k&16); vgpr = (k&15)>>1; half = k&1
    __shared__ __align__(32) _Float16 w2s[16][32][16];   // 16 KB (used once)
    __shared__ __align__(16) _Float16 ht[8][16][72];     // per-wave h tile (padded rows), 18 KB
    __shared__ float stg[8][16][8];                      // staged per-muon inputs
    __shared__ int   sbid[8][16];                        // staged batch indices
    __shared__ float lw1[7 * 64];
    __shared__ float lb1[64];
    __shared__ float lb2[128];

    const int tid = threadIdx.x;

    for (int i = tid; i < 7 * 64; i += 256) lw1[i] = w1[i];
    for (int i = tid; i < 64;     i += 256) lb1[i] = b1[i];
    for (int i = tid; i < 128;    i += 256) lb2[i] = b2[i];
    for (int idx = tid; idx < 64 * 128; idx += 256) {
        int k = idx >> 7, c = idx & 127;
        int frag = ((k >> 5) << 3) | (c >> 4);
        int lane = (c & 15) | (k & 16);
        int v    = (k & 15) >> 1;
        w2s[frag][lane][v * 2 + (k & 1)] = (_Float16)w2[idx];
    }
    __syncthreads();

    const int  wid  = tid >> 5;
    const int  lane = tid & 31;
    const int  m    = lane & 15;
    const bool hihalf = (lane & 16) != 0;

    // Hoist all 16 B fragments into registers for the whole kernel.
    v16h breg[16];
#pragma unroll
    for (int f = 0; f < 16; ++f) breg[f] = *(const v16h*)&w2s[f][lane][0];

    for (int r = 0; r < rounds; ++r) {
        const int  tile  = (r * (int)gridDim.x + (int)blockIdx.x) * 8 + wid;
        const bool valid = tile < nTiles;
        const int  base  = tile * 16;

        // ---- phase 1: stage inputs
        if (valid && lane < 16) {
            int mu = base + lane;
            float f0 = 0.f, f1 = 0.f, f2 = 0.f;
            int b = -1;
            if (mu < M) {
                f0 = muons[(long)mu * 3 + 0];
                f1 = muons[(long)mu * 3 + 1];
                f2 = muons[(long)mu * 3 + 2];
                b  = bidx[mu];
            }
            float c0 = 0.f, c1 = 0.f, c2 = 0.f, c3 = 0.f;
            if (b >= 0) {
                c0 = cond[(long)b * 4 + 0];
                c1 = cond[(long)b * 4 + 1];
                c2 = cond[(long)b * 4 + 2];
                c3 = cond[(long)b * 4 + 3];
            }
            stg[wid][lane][0] = f0; stg[wid][lane][1] = f1; stg[wid][lane][2] = f2;
            stg[wid][lane][3] = c0; stg[wid][lane][4] = c1; stg[wid][lane][5] = c2;
            stg[wid][lane][6] = c3;
            sbid[wid][lane] = b;
        }
        __syncthreads();

        // ---- phase 2: layer 1 (K=7, VALU), write f16 h-tile to LDS
        if (valid) {
            float in0 = stg[wid][m][0], in1 = stg[wid][m][1], in2 = stg[wid][m][2];
            float in3 = stg[wid][m][3], in4 = stg[wid][m][4], in5 = stg[wid][m][5];
            float in6 = stg[wid][m][6];
            const int jb = hihalf ? 32 : 0;
#pragma unroll 8
            for (int j = 0; j < 32; ++j) {
                int col = jb + j;
                float a = lb1[col];
                a += in0 * lw1[0 * 64 + col];
                a += in1 * lw1[1 * 64 + col];
                a += in2 * lw1[2 * 64 + col];
                a += in3 * lw1[3 * 64 + col];
                a += in4 * lw1[4 * 64 + col];
                a += in5 * lw1[5 * 64 + col];
                a += in6 * lw1[6 * 64 + col];
                ht[wid][m][col] = (_Float16)leaky02(a);
            }
        }
        __syncthreads();

        // ---- phase 3: layer 2 via WMMA (B already in registers), then fused
        //      segment-max scatter
        if (valid) {
            v8f acc[8];
#pragma unroll
            for (int n = 0; n < 8; ++n) acc[n] = (v8f)(0.0f);

#pragma unroll
            for (int ks = 0; ks < 2; ++ks) {
                const int koff = ks * 32 + (hihalf ? 16 : 0);
                // A fragment in contiguous k order (k = 16h + s), matching B: two aligned 16B LDS loads
                v8h lo = *(const v8h*)&ht[wid][m][koff];
                v8h hh = *(const v8h*)&ht[wid][m][koff + 8];
                v16h afrag = __builtin_shufflevector(lo, hh,
                    0, 1, 2, 3, 4, 5, 6, 7, 8, 9, 10, 11, 12, 13, 14, 15);
#pragma unroll
                for (int n = 0; n < 8; ++n) {
                    acc[n] = __builtin_amdgcn_wmma_f32_16x16x32_f16(
                        false, afrag, false, breg[ks * 8 + n], (short)0, acc[n],
                        false, false);
                }
            }

            const int rowoff = hihalf ? 8 : 0;
            const int b0 = sbid[wid][0];
            const int b15 = sbid[wid][15];
            const bool fast = (b0 == b15) && (b0 >= 0);   // whole tile = one segment

#pragma unroll
            for (int n = 0; n < 8; ++n) {
                const int c = n * 16 + m;
                const float bias = lb2[c];
                float vals[8];
#pragma unroll
                for (int rr = 0; rr < 8; ++rr) {
                    float x = leaky02(acc[n][rr] + bias);
                    // nan_to_num(nan->0, +-inf->0)
                    if (!(__builtin_fabsf(x) <= 3.0e38f)) x = 0.0f;
                    vals[rr] = x;
                }
                if (fast) {
                    float vmax = vals[0];
#pragma unroll
                    for (int rr = 1; rr < 8; ++rr) vmax = fmaxf(vmax, vals[rr]);
                    float o = __shfl_xor(vmax, 16, 32);
                    vmax = fmaxf(vmax, o);
                    if (!hihalf) atomic_fmax(&seg[(long)b0 * 128 + c], vmax);
                } else {
                    int curb = sbid[wid][rowoff];
                    float run = vals[0];
#pragma unroll
                    for (int rr = 1; rr < 8; ++rr) {
                        int bb = sbid[wid][rowoff + rr];
                        if (bb == curb) {
                            run = fmaxf(run, vals[rr]);
                        } else {
                            if (curb >= 0) atomic_fmax(&seg[(long)curb * 128 + c], run);
                            curb = bb; run = vals[rr];
                        }
                    }
                    if (curb >= 0) atomic_fmax(&seg[(long)curb * 128 + c], run);
                }
            }
        }
        __syncthreads();
    }
}

// ---------------------------------------------------------------- K2: decision net
// one block (128 threads) per event
__global__ __launch_bounds__(128) void decision_kernel(
    const float* __restrict__ seg,   // [B,128]
    const float* __restrict__ cond,  // [B,4]
    const float* __restrict__ w3,    // [132,128]
    const float* __restrict__ b3,    // [128]
    const float* __restrict__ w4,    // [128,1]
    const float* __restrict__ b4,    // [1]
    float*       __restrict__ out,   // [B]
    float inv_scale)
{
    __shared__ float gf[132];
    __shared__ float red[4];
    const int i = blockIdx.x;
    const int t = threadIdx.x;

    float x = seg[(long)i * 128 + t];
    x = fminf(fmaxf(x, -1.0e9f), 1.0e9f);   // nan_to_num bounds on global feats
    gf[t] = x;
    if (t < 4) gf[128 + t] = cond[(long)i * 4 + t];
    __syncthreads();

    float a = b3[t];
#pragma unroll 4
    for (int k = 0; k < 132; ++k) a += gf[k] * w3[k * 128 + t];
    a = leaky02(a);
    float p = a * w4[t];

#pragma unroll
    for (int off = 16; off > 0; off >>= 1) p += __shfl_down(p, off, 32);
    if ((t & 31) == 0) red[t >> 5] = p;
    __syncthreads();
    if (t == 0) {
        float s = red[0] + red[1] + red[2] + red[3] + b4[0];
        s *= inv_scale;
        s = fminf(fmaxf(s, -1000.0f), 1000.0f);
        out[i] = s;
    }
}

// ---------------------------------------------------------------- host
extern "C" void kernel_launch(void* const* d_in, const int* in_sizes, int n_in,
                              void* d_out, int out_size, void* d_ws, size_t ws_size,
                              hipStream_t stream) {
    (void)n_in; (void)out_size; (void)ws_size;
    const float* muons = (const float*)d_in[0];
    const int*   bidx  = (const int*)d_in[1];
    const float* cond  = (const float*)d_in[2];
    // d_in[3] = batch_size scalar (unused; derived from in_sizes)
    const float* w1 = (const float*)d_in[4];
    const float* b1 = (const float*)d_in[5];
    const float* w2 = (const float*)d_in[6];
    const float* b2 = (const float*)d_in[7];
    const float* w3 = (const float*)d_in[8];
    const float* b3 = (const float*)d_in[9];
    const float* w4 = (const float*)d_in[10];
    const float* b4 = (const float*)d_in[11];

    const int M = in_sizes[0] / 3;
    const int B = in_sizes[2] / 4;

    float* seg = (float*)d_ws;              // [B,128] scratch, 8 MB @ B=16384
    float* out = (float*)d_out;

    const int segN = B * 128;
    hipLaunchKernelGGL(seg_init_kernel, dim3((segN + 255) / 256), dim3(256), 0, stream,
                       seg, segN);

    const int nTiles = (M + 15) / 16;
    int grid = (nTiles + 7) / 8;
    if (grid > 2048) grid = 2048;
    const int rounds = (nTiles + grid * 8 - 1) / (grid * 8);
    hipLaunchKernelGGL(point_kernel, dim3(grid), dim3(256), 0, stream,
                       muons, bidx, cond, w1, b1, w2, b2, seg, M, nTiles, rounds);

    const float inv_scale = 1.0f / fmaxf(1.0f, sqrtf((float)B));
    hipLaunchKernelGGL(decision_kernel, dim3(B), dim3(128), 0, stream,
                       seg, cond, w3, b3, w4, b4, out, inv_scale);
}
